// PointNet2MSG_45414984188393
// MI455X (gfx1250) — matrix operand order repaired and checked by reading, not compiled, not run on hardware
//
#include <hip/hip_runtime.h>

typedef __attribute__((ext_vector_type(16))) _Float16 v16h;
typedef __attribute__((ext_vector_type(8)))  _Float16 v8h;
typedef __attribute__((ext_vector_type(8)))  float    v8f;

#define BT   16
#define NPTS 1024

// ---------------------------------------------------------------------------
// Input transpose: xyz (1,16,3,1024) -> pts [16][1024][3]
// ---------------------------------------------------------------------------
__global__ void k_txyz(const float* __restrict__ in, float* __restrict__ pts) {
  int e = blockIdx.x * blockDim.x + threadIdx.x;
  if (e >= BT * NPTS * 3) return;
  int b = e / (NPTS * 3), r = e % (NPTS * 3), p = r / 3, c = r % 3;
  pts[e] = in[((size_t)b * 3 + c) * NPTS + p];
}

// ---------------------------------------------------------------------------
// Per-layer prep: W f32 [Cout,Cin] -> f16 [Cout,Kpad] (zero-padded K),
// fold conv-bias + BatchNorm (eval) into per-channel scale/shift.
// ---------------------------------------------------------------------------
__global__ void k_prep(const float* __restrict__ W, const float* __restrict__ bias,
                       const float* __restrict__ g, const float* __restrict__ bt,
                       const float* __restrict__ mean, const float* __restrict__ var,
                       int Cin, int K, int Cout,
                       _Float16* __restrict__ Wh, float* __restrict__ scale,
                       float* __restrict__ shift) {
  int e = blockIdx.x * blockDim.x + threadIdx.x;
  if (e >= Cout * K) return;
  int co = e / K, k = e % K;
  Wh[e] = (k < Cin) ? (_Float16)W[(size_t)co * Cin + k] : (_Float16)0.f;
  if (k == 0) {
    float s = g[co] * rsqrtf(var[co] + 1e-5f);
    scale[co] = s;
    shift[co] = (bias[co] - mean[co]) * s + bt[co];
  }
}

// ---------------------------------------------------------------------------
// Farthest point sampling: one block per batch, one thread per point.
// Emits the selected centroid coordinates directly (idx[0]=0 semantics of
// the reference scan: record current 'far' before the argmax update).
// ---------------------------------------------------------------------------
__global__ __launch_bounds__(1024) void k_fps(const float* __restrict__ pts,
                                              float* __restrict__ nx,
                                              int N, int S) {
  __shared__ float sval[1024];
  __shared__ int   sidx[1024];
  __shared__ float cc[3];
  int b = blockIdx.x, t = threadIdx.x;
  const float* P = pts + (size_t)b * N * 3;
  float x = P[t * 3], y = P[t * 3 + 1], z = P[t * 3 + 2];
  float dist = 1e10f;
  int sel = 0;
  for (int i = 0; i < S; ++i) {
    if (t == 0) {
      cc[0] = P[sel * 3]; cc[1] = P[sel * 3 + 1]; cc[2] = P[sel * 3 + 2];
      float* o = nx + ((size_t)b * S + i) * 3;
      o[0] = cc[0]; o[1] = cc[1]; o[2] = cc[2];
    }
    __syncthreads();
    float dx = x - cc[0], dy = y - cc[1], dz = z - cc[2];
    dist = fminf(dist, dx * dx + dy * dy + dz * dz);
    sval[t] = dist; sidx[t] = t;
    __syncthreads();
    for (int off = N >> 1; off > 0; off >>= 1) {
      if (t < off) {
        float vr = sval[t + off]; int ir = sidx[t + off];
        if (vr > sval[t] || (vr == sval[t] && ir < sidx[t])) { sval[t] = vr; sidx[t] = ir; }
      }
      __syncthreads();
    }
    sel = sidx[0];
  }
}

// ---------------------------------------------------------------------------
// Ball query: first <=ns hits in index order (== reference sort+slice),
// pad misses with first hit. One thread per centroid.
// ---------------------------------------------------------------------------
__global__ void k_ballquery(const float* __restrict__ pts, const float* __restrict__ cents,
                            int N, int S, float r2, int ns, int* __restrict__ idx, int b) {
  int s = blockIdx.x * blockDim.x + threadIdx.x;
  if (s >= S) return;
  const float* c = cents + ((size_t)b * S + s) * 3;
  float cx = c[0], cy = c[1], cz = c[2];
  const float* P = pts + (size_t)b * N * 3;
  int cnt = 0, first = 0;
  for (int p = 0; p < N && cnt < ns; ++p) {
    float dx = P[p * 3] - cx, dy = P[p * 3 + 1] - cy, dz = P[p * 3 + 2] - cz;
    float d = dx * dx + dy * dy + dz * dz;
    if (d <= r2) { if (cnt == 0) first = p; idx[s * ns + cnt] = p; ++cnt; }
  }
  for (int j = cnt; j < ns; ++j) idx[s * ns + j] = first;
}

// ---------------------------------------------------------------------------
// Gather grouped MLP input, f16, zero-padded to K columns.
// Layout per row: [feat(featC) | rel_xyz(3) | zero-pad]   (featC may be 0)
// ---------------------------------------------------------------------------
__global__ void k_group(const float* __restrict__ pts, const float* __restrict__ feat,
                        int featC, const float* __restrict__ cents,
                        const int* __restrict__ idx, int b,
                        int N, int S, int ns, int K, _Float16* __restrict__ X) {
  int e = blockIdx.x * blockDim.x + threadIdx.x;
  int total = S * ns * K;
  if (e >= total) return;
  int row = e / K, k = e % K;
  int s = row / ns, j = row % ns;
  int p = idx[s * ns + j];
  float val;
  if (k < featC) {
    val = feat[((size_t)b * N + p) * featC + k];
  } else if (k < featC + 3) {
    int c = k - featC;
    val = pts[((size_t)b * N + p) * 3 + c] - cents[((size_t)b * S + s) * 3 + c];
  } else {
    val = 0.f;
  }
  X[(size_t)row * K + k] = (_Float16)val;
}

// ---------------------------------------------------------------------------
// SA3 group-all: X[2048][672] = [xyz2(3) | l2_feat(640) | pad]
// ---------------------------------------------------------------------------
__global__ void k_groupall(const float* __restrict__ xyz2, const float* __restrict__ feat2,
                           _Float16* __restrict__ X) {
  const int K = 672;
  int e = blockIdx.x * blockDim.x + threadIdx.x;
  if (e >= 2048 * K) return;
  int row = e / K, k = e % K;
  float val;
  if (k < 3)        val = xyz2[(size_t)row * 3 + k];
  else if (k < 643) val = feat2[(size_t)row * 640 + (k - 3)];
  else              val = 0.f;
  X[(size_t)row * K + k] = (_Float16)val;
}

// ---------------------------------------------------------------------------
// WMMA GEMM with fused BN+ReLU epilogue.
// Y[M,Cout] = act( (X[M,K] . W[Cout,K]^T) * scale[c] + shift[c] )
// 8 waves/block, one 16x16 tile per wave, K-loop in steps of 32.
// M, Cout multiples of 16; K multiple of 32; rows 16B-aligned.
// ---------------------------------------------------------------------------
__global__ __launch_bounds__(256) void k_gemm(
    const _Float16* __restrict__ X, const _Float16* __restrict__ W,
    const float* __restrict__ scale, const float* __restrict__ shift,
    _Float16* __restrict__ Y, int M, int K, int Cout, int relu) {
  const int wid  = threadIdx.x >> 5;
  const int lane = threadIdx.x & 31;
  const int nT = Cout >> 4, mT = M >> 4;
  int tile = blockIdx.x * 8 + wid;
  if (tile >= mT * nT) return;                 // wave-uniform: EXEC stays full
  const int mtile = tile / nT, ntile = tile % nT;
  const int sub = lane & 15, hi = lane >> 4;

  const _Float16* arow = X + (size_t)(mtile * 16 + sub) * K + hi * 8;
  const _Float16* brow = W + (size_t)(ntile * 16 + sub) * K + hi * 8;
  v8f acc = {};
  for (int k0 = 0; k0 < K; k0 += 32) {
    v8h a0 = *(const v8h*)(arow + k0);          // K = kb..kb+7
    v8h a1 = *(const v8h*)(arow + k0 + 16);     // K = 16+kb..16+kb+7
    v16h a = __builtin_shufflevector(a0, a1, 0,1,2,3,4,5,6,7,8,9,10,11,12,13,14,15);
    v16h bf = __builtin_shufflevector(*(const v8h*)(brow + k0), *(const v8h*)(brow + k0 + 16), 0,1,2,3,4,5,6,7,8,9,10,11,12,13,14,15);  // K = kb..kb+7, 16+kb..16+kb+7
    acc = __builtin_amdgcn_wmma_f32_16x16x32_f16(false, a, false, bf,
                                                 (short)0, acc, false, false);
  }
  const int n = ntile * 16 + sub;
  const float sc = scale[n], sh = shift[n];
  _Float16* ycol = Y + n;
#pragma unroll
  for (int r = 0; r < 8; ++r) {
    int m = mtile * 16 + r + hi * 8;            // C/D layout: M = r + hi*8, N = sub
    float y = acc[r] * sc + sh;
    if (relu) y = y > 0.f ? y : 0.f;
    ycol[(size_t)m * Cout] = (_Float16)y;
  }
}

// ---------------------------------------------------------------------------
// Max-pool over ns neighbors -> f32 slice of the concatenated feature map.
// ---------------------------------------------------------------------------
__global__ void k_maxpool(const _Float16* __restrict__ Y, int S, int ns, int C,
                          float* __restrict__ out, int ostride) {
  int e = blockIdx.x * blockDim.x + threadIdx.x;
  if (e >= S * C) return;
  int s = e / C, c = e % C;
  const _Float16* p = Y + (size_t)s * ns * C + c;
  float m = -3.0e38f;
  for (int j = 0; j < ns; ++j) { float v = (float)p[(size_t)j * C]; m = v > m ? v : m; }
  out[(size_t)s * ostride + c] = m;
}

// SA3 pool: max over 128 points per batch -> l3 (f32 into d_out) + f16 for FC.
__global__ void k_pool3(const _Float16* __restrict__ Y, float* __restrict__ l3,
                        _Float16* __restrict__ xf16) {
  int e = blockIdx.x * blockDim.x + threadIdx.x;
  if (e >= BT * 1024) return;
  int b = e >> 10, c = e & 1023;
  const _Float16* p = Y + ((size_t)b * 128) * 1024 + c;
  float m = -3.0e38f;
  for (int s = 0; s < 128; ++s) { float v = (float)p[(size_t)s * 1024]; m = v > m ? v : m; }
  l3[e] = m;
  xf16[e] = (_Float16)m;
}

// ---------------------------------------------------------------------------
// FC3 (256->14) + log_softmax + transpose to pred[c*16+t]. One block.
// ---------------------------------------------------------------------------
__global__ void k_head(const _Float16* __restrict__ X, const float* __restrict__ W3,
                       const float* __restrict__ b3, float* __restrict__ out) {
  __shared__ float lg[16][14];
  int tid = threadIdx.x;
  int t = tid / 14, c = tid % 14;
  if (tid < 224) {
    float acc = b3[c];
    for (int k = 0; k < 256; ++k) acc += (float)X[t * 256 + k] * W3[c * 256 + k];
    lg[t][c] = acc;
  }
  __syncthreads();
  if (tid < 224) {
    float mx = -3.0e38f;
    for (int i = 0; i < 14; ++i) mx = fmaxf(mx, lg[t][i]);
    float se = 0.f;
    for (int i = 0; i < 14; ++i) se += expf(lg[t][i] - mx);
    out[c * 16 + t] = lg[t][c] - mx - logf(se);
  }
}

// ---------------------------------------------------------------------------
// Host orchestration
// ---------------------------------------------------------------------------
extern "C" void kernel_launch(void* const* d_in, const int* in_sizes, int n_in,
                              void* d_out, int out_size, void* d_ws, size_t ws_size,
                              hipStream_t stream) {
  (void)in_sizes; (void)n_in; (void)out_size; (void)ws_size;
  const float* xyz = (const float*)d_in[0];
  float* outp = (float*)d_out;

  char* ws = (char*)d_ws;
  size_t cur = 0;
  auto alloc = [&](size_t bytes) -> char* {
    cur = (cur + 255) & ~(size_t)255;
    char* p = ws + cur;
    cur += bytes;
    return p;
  };

  float* pts  = (float*)alloc((size_t)BT * NPTS * 3 * 4);
  float* nx1  = (float*)alloc((size_t)BT * 512 * 3 * 4);
  float* nx2  = (float*)alloc((size_t)BT * 128 * 3 * 4);
  float* l1f  = (float*)alloc((size_t)BT * 512 * 320 * 4);   // SA1 concat feats
  float* l2f  = (float*)alloc((size_t)BT * 128 * 640 * 4);   // SA2 concat feats
  int*   idxb = (int*)  alloc((size_t)512 * 128 * 4);        // per-batch idx scratch
  _Float16* ping = (_Float16*)alloc((size_t)6291456 * 2);    // max 65536x96 etc.
  _Float16* pong = (_Float16*)alloc((size_t)8388608 * 2);    // max 65536x128

  // ----- layer table (d_in flattening: insertion order; each layer = W,b,g,bt,m,v)
  struct LD { int base, Cin, Cout, K; };
  LD L[23];
  _Float16* Wh[23]; float* Sc[23]; float* Sh[23];
  {
    int li = 0, in = 1;
    const int sa1c[3][4] = {{3,32,32,64},{3,64,64,128},{3,64,96,128}};
    for (int br = 0; br < 3; ++br)
      for (int l = 0; l < 3; ++l) { L[li] = {in, sa1c[br][l], sa1c[br][l+1], (l==0)?32:sa1c[br][l]}; in += 6; ++li; }
    const int sa2c[3][4] = {{323,64,64,128},{323,128,128,256},{323,128,128,256}};
    for (int br = 0; br < 3; ++br)
      for (int l = 0; l < 3; ++l) { L[li] = {in, sa2c[br][l], sa2c[br][l+1], (l==0)?352:sa2c[br][l]}; in += 6; ++li; }
    const int sa3c[4] = {643,256,512,1024};
    for (int l = 0; l < 3; ++l) { L[li] = {in, sa3c[l], sa3c[l+1], (l==0)?672:sa3c[l]}; in += 6; ++li; }
    L[li++] = {in, 1024, 512, 1024}; in += 6;   // fc1 + bn1 (same W,b,g,bt,m,v order)
    L[li++] = {in, 512, 256, 512};   in += 6;   // fc2 + bn2
  }
  for (int i = 0; i < 23; ++i) {
    Wh[i] = (_Float16*)alloc((size_t)L[i].Cout * L[i].K * 2);
    Sc[i] = (float*)alloc((size_t)L[i].Cout * 4);
    Sh[i] = (float*)alloc((size_t)L[i].Cout * 4);
  }
  const float* W3 = (const float*)d_in[139];
  const float* b3 = (const float*)d_in[140];

  // ----- input transpose + weight prep
  { int n = BT * NPTS * 3; k_txyz<<<(n + 255) / 256, 256, 0, stream>>>(xyz, pts); }
  for (int i = 0; i < 23; ++i) {
    int n = L[i].Cout * L[i].K;
    k_prep<<<(n + 255) / 256, 256, 0, stream>>>(
        (const float*)d_in[L[i].base],     (const float*)d_in[L[i].base + 1],
        (const float*)d_in[L[i].base + 2], (const float*)d_in[L[i].base + 3],
        (const float*)d_in[L[i].base + 4], (const float*)d_in[L[i].base + 5],
        L[i].Cin, L[i].K, L[i].Cout, Wh[i], Sc[i], Sh[i]);
  }

  // ----- SA1
  k_fps<<<BT, NPTS, 0, stream>>>(pts, nx1, NPTS, 512);
  {
    const int   ns[3]    = {16, 32, 128};
    const float rad[3]   = {0.1f, 0.2f, 0.4f};
    const int   choff[3] = {0, 64, 192};
    for (int br = 0; br < 3; ++br) {
      for (int b = 0; b < BT; ++b) {
        k_ballquery<<<2, 256, 0, stream>>>(pts, nx1, NPTS, 512, rad[br]*rad[br], ns[br], idxb, b);
        int rows = 512 * ns[br];
        { int n = rows * 32;
          k_group<<<(n + 255) / 256, 256, 0, stream>>>(pts, nullptr, 0, nx1, idxb, b,
                                                       NPTS, 512, ns[br], 32, ping); }
        _Float16* bufs[4] = {ping, pong, ping, pong};
        for (int l = 0; l < 3; ++l) {
          int li = br * 3 + l;
          int tiles = (rows / 16) * (L[li].Cout / 16);
          k_gemm<<<(tiles + 7) / 8, 256, 0, stream>>>(bufs[l], Wh[li], Sc[li], Sh[li],
                                                      bufs[l + 1], rows, L[li].K, L[li].Cout, 1);
        }
        int C = L[br * 3 + 2].Cout;
        { int n = 512 * C;
          k_maxpool<<<(n + 255) / 256, 256, 0, stream>>>(
              pong, 512, ns[br], C, l1f + (size_t)b * 512 * 320 + choff[br], 320); }
      }
    }
  }

  // ----- SA2
  k_fps<<<BT, 512, 0, stream>>>(nx1, nx2, 512, 128);
  {
    const int   ns[3]    = {32, 64, 128};
    const float rad[3]   = {0.2f, 0.4f, 0.8f};
    const int   choff[3] = {0, 128, 384};
    for (int br = 0; br < 3; ++br) {
      for (int b = 0; b < BT; ++b) {
        k_ballquery<<<1, 256, 0, stream>>>(nx1, nx2, 512, 128, rad[br]*rad[br], ns[br], idxb, b);
        int rows = 128 * ns[br];
        { int n = rows * 352;
          k_group<<<(n + 255) / 256, 256, 0, stream>>>(nx1, l1f, 320, nx2, idxb, b,
                                                       512, 128, ns[br], 352, ping); }
        _Float16* bufs[4] = {ping, pong, ping, pong};
        for (int l = 0; l < 3; ++l) {
          int li = 9 + br * 3 + l;
          int tiles = (rows / 16) * (L[li].Cout / 16);
          k_gemm<<<(tiles + 7) / 8, 256, 0, stream>>>(bufs[l], Wh[li], Sc[li], Sh[li],
                                                      bufs[l + 1], rows, L[li].K, L[li].Cout, 1);
        }
        int C = L[9 + br * 3 + 2].Cout;
        { int n = 128 * C;
          k_maxpool<<<(n + 255) / 256, 256, 0, stream>>>(
              pong, 128, ns[br], C, l2f + (size_t)b * 128 * 640 + choff[br], 640); }
      }
    }
  }

  // ----- SA3 (group-all, all batches at once: M = 2048)
  { int n = 2048 * 672; k_groupall<<<(n + 255) / 256, 256, 0, stream>>>(nx2, l2f, ping); }
  {
    _Float16* bufs[4] = {ping, pong, ping, pong};
    for (int l = 0; l < 3; ++l) {
      int li = 18 + l;
      int tiles = (2048 / 16) * (L[li].Cout / 16);
      k_gemm<<<(tiles + 7) / 8, 256, 0, stream>>>(bufs[l], Wh[li], Sc[li], Sh[li],
                                                  bufs[l + 1], 2048, L[li].K, L[li].Cout, 1);
    }
  }
  { int n = BT * 1024; k_pool3<<<(n + 255) / 256, 256, 0, stream>>>(pong, outp + 224, ping); }

  // ----- FC head (M = 16)
  k_gemm<<<(32 + 7) / 8, 256, 0, stream>>>(ping, Wh[21], Sc[21], Sh[21], pong, 16, 1024, 512, 1);
  k_gemm<<<(16 + 7) / 8, 256, 0, stream>>>(pong, Wh[22], Sc[22], Sh[22], ping, 16, 512, 256, 1);
  k_head<<<1, 224, 0, stream>>>(ping, W3, b3, outp);
}
